// Attention_39522289058137
// MI455X (gfx1250) — compile-verified, matrix operand order checked by reading, not run on hardware
//
#include <hip/hip_runtime.h>
#include <hip/hip_bf16.h>

typedef __attribute__((ext_vector_type(16))) __bf16 bf16x16;
typedef __attribute__((ext_vector_type(8)))  __bf16 bf16x8;
typedef __attribute__((ext_vector_type(4)))  __bf16 bf16x4;
typedef __attribute__((ext_vector_type(8)))  float  f32x8;
typedef __attribute__((ext_vector_type(4)))  float  f32x4;

#define EMBED 768
#define SEQ   1024
#define BATCH 32

#define F32X8_ZERO {0.f,0.f,0.f,0.f,0.f,0.f,0.f,0.f}

// Load a 16-element bf16 WMMA operand fragment for this lane (global or LDS).
// CDNA5 16-bit 16x32 A/B layout (wave32): lane group h (=lane>>4) holds K
// elements {k0+8h..+7} in VGPRs 0-3 and {k0+16+8h..+7} in VGPRs 4-7:
// two contiguous b128 loads.
__device__ inline bf16x16 load_frag(const __bf16* p /* = row_base + k0 + 8*h */) {
    union { bf16x16 v; bf16x8 h2[2]; } u;
    u.h2[0] = *(const bf16x8*)(p);
    u.h2[1] = *(const bf16x8*)(p + 16);
    return u.v;
}

// gfx1250 async copy: 16 bytes global -> LDS, tracked by ASYNCcnt
// (CDNA5 ISA: GLOBAL_LOAD_ASYNC_TO_LDS_B128,
//  VDST = per-lane LDS byte offset, VADDR = per-lane 64-bit global address).
__device__ inline void async_copy16(unsigned lds_byte, const __bf16* g) {
    asm volatile("global_load_async_to_lds_b128 %0, %1, off"
                 :: "v"(lds_byte), "v"(g) : "memory");
}
__device__ inline void wait_async0() {
    asm volatile("s_wait_asynccnt 0x0" ::: "memory");
}

// ---------------------------------------------------------------------------
// fp32 -> bf16 pack (vectorized, grid-stride)
// ---------------------------------------------------------------------------
__global__ void cvt_bf16_kernel(const float* __restrict__ src,
                                __bf16* __restrict__ dst, int n4) {
    int i = blockIdx.x * blockDim.x + threadIdx.x;
    int stride = gridDim.x * blockDim.x;
    for (; i < n4; i += stride) {
        f32x4 v = ((const f32x4*)src)[i];
        bf16x4 o;
        o[0] = (__bf16)v[0]; o[1] = (__bf16)v[1];
        o[2] = (__bf16)v[2]; o[3] = (__bf16)v[3];
        ((bf16x4*)dst)[i] = o;
    }
}

// ---------------------------------------------------------------------------
// QKV projection: [32768 x 768] x [768 x 768]^T + bias for Q, K, V.
// Each wave computes a 32x32 output macro-tile (2x2 WMMA tiles): per k-step
// 8 b128 loads feed 4 v_wmma_f32_16x16x32_bf16 (2x the L2 arithmetic
// intensity of a single-tile wave). Q,K stored bf16 [BN,O]; V stored bf16
// transposed [B][O][SEQ] so the attention P*V B-fragment is contiguous.
// ---------------------------------------------------------------------------
__global__ __launch_bounds__(256)
void qkv_kernel(const __bf16* __restrict__ xb, const __bf16* __restrict__ Wb,
                const float* __restrict__ bq, const float* __restrict__ bk,
                const float* __restrict__ bv,
                __bf16* __restrict__ Qb, __bf16* __restrict__ Kb,
                __bf16* __restrict__ Vt) {
    const int lane = threadIdx.x & 31;
    const int l15  = lane & 15;
    const int h    = lane >> 4;
    const int wid  = threadIdx.x >> 5;
    const int wm   = wid >> 1;                    // 0..3
    const int wn   = wid & 1;                     // 0..1

    const int m0  = blockIdx.x * 128 + wm * 32;   // token rows (32)
    const int n0  = blockIdx.y * 64 + wn * 32;    // output features (32)
    const int mat = blockIdx.z;                   // 0=Q 1=K 2=V

    const __bf16* W    = Wb + (size_t)mat * EMBED * EMBED;
    const float*  bias = (mat == 0) ? bq : ((mat == 1) ? bk : bv);

    const __bf16* ap0 = xb + (size_t)(m0 + l15) * EMBED + 8 * h;
    const __bf16* ap1 = ap0 + 16 * EMBED;
    const __bf16* bp0 = W  + (size_t)(n0 + l15) * EMBED + 8 * h;
    const __bf16* bp1 = bp0 + 16 * EMBED;

    f32x8 acc[2][2];
    for (int i = 0; i < 2; ++i)
        for (int j = 0; j < 2; ++j) acc[i][j] = F32X8_ZERO;

    for (int k0 = 0; k0 < EMBED; k0 += 32) {
        bf16x16 a0 = load_frag(ap0 + k0);
        bf16x16 a1 = load_frag(ap1 + k0);
        bf16x16 b0 = load_frag(bp0 + k0);
        bf16x16 b1 = load_frag(bp1 + k0);
        acc[0][0] = __builtin_amdgcn_wmma_f32_16x16x32_bf16(
                        false, a0, false, b0, (short)0, acc[0][0], false, false);
        acc[0][1] = __builtin_amdgcn_wmma_f32_16x16x32_bf16(
                        false, a0, false, b1, (short)0, acc[0][1], false, false);
        acc[1][0] = __builtin_amdgcn_wmma_f32_16x16x32_bf16(
                        false, a1, false, b0, (short)0, acc[1][0], false, false);
        acc[1][1] = __builtin_amdgcn_wmma_f32_16x16x32_bf16(
                        false, a1, false, b1, (short)0, acc[1][1], false, false);
    }

    // C layout: lane column n = lane&15; VGPR r holds row m = r + 8*(lane>>4)
    for (int sm = 0; sm < 2; ++sm) {
        for (int sn = 0; sn < 2; ++sn) {
            const int mt = m0 + sm * 16;
            const int nt = n0 + sn * 16;
            const float bval = bias[nt + l15];
            if (mat < 2) {
                __bf16* outp = (mat == 0) ? Qb : Kb;
                for (int r = 0; r < 8; ++r) {
                    int m = mt + r + 8 * h;
                    outp[(size_t)m * EMBED + (nt + l15)] =
                        (__bf16)(acc[sm][sn][r] + bval);
                }
            } else {
                // Vt[b][o][token]: rows r are contiguous tokens at fixed o.
                int b    = mt / SEQ;
                int mloc = (mt % SEQ) + 8 * h;
                bf16x8 pk;
                for (int r = 0; r < 8; ++r)
                    pk[r] = (__bf16)(acc[sm][sn][r] + bval);
                *(bf16x8*)(Vt + (size_t)b * EMBED * SEQ +
                           (size_t)(nt + l15) * SEQ + mloc) = pk;
            }
        }
    }
}

// ---------------------------------------------------------------------------
// Flash attention. One block = (batch b, 32 q rows). 8 waves: wid>>2 picks
// the 16-row q tile, wid&3 picks a 192-wide o chunk (12 fp32 accum tiles).
// Per 32-key step: the 32x768 bf16 K tile is staged ONCE per block into LDS
// with global_load_async_to_lds_b128 (ASYNCcnt) and shared by all 8 waves;
// S = Q*K^T reads K from LDS; online softmax via shfl_xor in 16-lane halves;
// P transposed through a per-wave LDS tile into an A-fragment; 12 WMMAs
// accumulate P*V from the transposed V.
// ---------------------------------------------------------------------------
#define KTILE_E (32 * EMBED)            // K stage region, elements
#define PTILE_E 512                     // per-wave 16x32 P tile, elements

__global__ __launch_bounds__(256)
void attn_kernel(const __bf16* __restrict__ Qb, const __bf16* __restrict__ Kb,
                 const __bf16* __restrict__ Vt, float* __restrict__ out) {
    __shared__ __bf16 smem[KTILE_E + 8 * PTILE_E];   // 48KB + 8KB

    const int lane = threadIdx.x & 31;
    const int l15  = lane & 15;
    const int h    = lane >> 4;
    const int wid  = threadIdx.x >> 5;

    const int b    = blockIdx.x >> 5;        // batch
    const int qp   = blockIdx.x & 31;        // 32-row q pair index
    const int qsel = wid >> 2;               // which 16-row q tile
    const int oc   = wid & 3;                // o chunk: [oc*192, oc*192+192)

    const unsigned smem_base = (unsigned)(unsigned long long)(void*)smem;
    const float inv_scale = 1.0f / (sqrtf((float)EMBED) + 1e-6f);

    const __bf16* qpiv =
        Qb + (size_t)(b * SEQ + qp * 32 + qsel * 16 + l15) * EMBED + 8 * h;
    const __bf16* kgb = Kb + (size_t)(b * SEQ) * EMBED;
    const __bf16* vtb = Vt + (size_t)b * EMBED * SEQ;
    __bf16* pl = smem + KTILE_E + wid * PTILE_E;

    f32x8 acc[12];
    for (int t = 0; t < 12; ++t) acc[t] = F32X8_ZERO;
    float mrow[8], lrow[8];
    for (int r = 0; r < 8; ++r) { mrow[r] = -1e30f; lrow[r] = 0.0f; }

    for (int j0 = 0; j0 < SEQ; j0 += 32) {
        // ---- cooperative async stage of 32x768 K tile into LDS ----
        __syncthreads();                       // prior reads of K tile done
        {
            const __bf16* gk = kgb + (size_t)j0 * EMBED;
            for (int i = 0; i < 12; ++i) {     // 3072 x 16B / 256 threads
                int c = threadIdx.x + 256 * i;
                async_copy16(smem_base + c * 16, gk + c * 8);
            }
        }
        wait_async0();
        __syncthreads();                       // K tile visible to all waves

        // ---- S = Q * K^T for 32 keys (two 16x16 tiles), K from LDS ----
        f32x8 s[2];
        for (int t = 0; t < 2; ++t) {
            f32x8 st = F32X8_ZERO;
            const __bf16* kld = smem + (t * 16 + l15) * EMBED + 8 * h;
            for (int k0 = 0; k0 < EMBED; k0 += 32) {
                bf16x16 a  = load_frag(qpiv + k0);
                bf16x16 bb = load_frag(kld + k0);
                st = __builtin_amdgcn_wmma_f32_16x16x32_bf16(
                         false, a, false, bb, (short)0, st, false, false);
            }
            s[t] = st;
        }

        // ---- online softmax (rows of C layout live in 16-lane halves) ----
        float p0[8], p1[8], alpha[8];
        for (int r = 0; r < 8; ++r) {
            float s0 = s[0][r] * inv_scale;
            float s1 = s[1][r] * inv_scale;
            float rm = fmaxf(s0, s1);
            rm = fmaxf(rm, __shfl_xor(rm, 1, 32));
            rm = fmaxf(rm, __shfl_xor(rm, 2, 32));
            rm = fmaxf(rm, __shfl_xor(rm, 4, 32));
            rm = fmaxf(rm, __shfl_xor(rm, 8, 32));
            float mnew = fmaxf(mrow[r], rm);
            alpha[r] = __expf(mrow[r] - mnew);
            mrow[r]  = mnew;
            p0[r] = __expf(s0 - mnew);
            p1[r] = __expf(s1 - mnew);
            float ps = p0[r] + p1[r];
            ps += __shfl_xor(ps, 1, 32);
            ps += __shfl_xor(ps, 2, 32);
            ps += __shfl_xor(ps, 4, 32);
            ps += __shfl_xor(ps, 8, 32);
            lrow[r] = alpha[r] * lrow[r] + ps;
        }

        for (int t = 0; t < 12; ++t)
            for (int r = 0; r < 8; ++r)
                acc[t][r] *= alpha[r];

        // ---- C-layout -> A-fragment transpose of P through LDS ----
        for (int r = 0; r < 8; ++r) {
            int m = r + 8 * h;
            pl[m * 32 + l15]      = (__bf16)p0[r];
            pl[m * 32 + 16 + l15] = (__bf16)p1[r];
        }
        bf16x16 pfrag;
        {
            const __bf16* pp = pl + l15 * 32 + 8 * h;
            union { bf16x16 v; bf16x8 h2[2]; } u;
            u.h2[0] = *(const bf16x8*)(pp);
            u.h2[1] = *(const bf16x8*)(pp + 16);
            pfrag = u.v;
        }

        // ---- O += P * V over this wave's 192-wide o chunk ----
        for (int t = 0; t < 12; ++t) {
            const __bf16* vp =
                vtb + (size_t)(oc * 192 + t * 16 + l15) * SEQ + j0 + 8 * h;
            bf16x16 bfr = load_frag(vp);
            acc[t] = __builtin_amdgcn_wmma_f32_16x16x32_bf16(
                         false, pfrag, false, bfr, (short)0, acc[t], false, false);
        }
    }

    // ---- epilogue: normalize and store fp32 ----
    float rl[8];
    for (int r = 0; r < 8; ++r) rl[r] = 1.0f / lrow[r];
    for (int t = 0; t < 12; ++t)
        for (int r = 0; r < 8; ++r) {
            int q = qp * 32 + qsel * 16 + r + 8 * h;
            int o = oc * 192 + t * 16 + l15;
            out[(size_t)(b * SEQ + q) * EMBED + o] = acc[t][r] * rl[r];
        }
}

// ---------------------------------------------------------------------------
extern "C" void kernel_launch(void* const* d_in, const int* in_sizes, int n_in,
                              void* d_out, int out_size, void* d_ws, size_t ws_size,
                              hipStream_t stream) {
    const float* x  = (const float*)d_in[0];
    const float* Wq = (const float*)d_in[1];
    const float* bq = (const float*)d_in[2];
    const float* Wk = (const float*)d_in[3];
    const float* bk = (const float*)d_in[4];
    const float* Wv = (const float*)d_in[5];
    const float* bv = (const float*)d_in[6];
    float* out = (float*)d_out;

    const size_t BN = (size_t)BATCH * SEQ;          // 32768
    const size_t XE = BN * EMBED;                   // 25,165,824 elems
    const size_t WE = (size_t)EMBED * EMBED;        // 589,824 elems

    __bf16* xb = (__bf16*)d_ws;                     // [BN, 768]
    __bf16* Wb = xb + XE;                           // [3][768,768]
    __bf16* Qb = Wb + 3 * WE;                       // [BN, 768]
    __bf16* Kb = Qb + XE;                           // [BN, 768]
    __bf16* Vt = Kb + XE;                           // [B][768][1024]

    // 1) pack fp32 -> bf16
    cvt_bf16_kernel<<<4096, 256, 0, stream>>>(x, xb, (int)(XE / 4));
    cvt_bf16_kernel<<<576, 256, 0, stream>>>(Wq, Wb + 0 * WE, (int)(WE / 4));
    cvt_bf16_kernel<<<576, 256, 0, stream>>>(Wk, Wb + 1 * WE, (int)(WE / 4));
    cvt_bf16_kernel<<<576, 256, 0, stream>>>(Wv, Wb + 2 * WE, (int)(WE / 4));

    // 2) QKV projections (Q,K row-major; V transposed), bias fused
    dim3 qkv_grid(256, 12, 3);   // M/128, O/64, {Q,K,V}
    qkv_kernel<<<qkv_grid, 256, 0, stream>>>(xb, Wb, bq, bk, bv, Qb, Kb, Vt);

    // 3) flash attention: one block per (batch, 32 q rows)
    attn_kernel<<<BATCH * 32, 256, 0, stream>>>(Qb, Kb, Vt, out);
}
